// GATReasoningScorer_9740985827546
// MI455X (gfx1250) — matrix operand order repaired and checked by reading, not compiled, not run on hardware
//
#include <hip/hip_runtime.h>
#include <hip/hip_bf16.h>
#include <cstdint>

// ---------------- problem constants (match reference) ----------------
#define NN 50000            // nodes
#define CC 128              // channels
#define EE 800000           // edges (before self loops)
#define PP 8192             // pos/neg pairs
#define ETOT (EE + NN)      // edges incl. self loops
#define NEG_SLOPE 0.2f

#define STRD 136            // LDS row stride (bf16) for K=128 images (+8 pad)
#define STRS 264            // LDS row stride (bf16) for K=256 images (+8 pad)

typedef __bf16 bf16_t;
typedef __bf16 v16bf __attribute__((ext_vector_type(16)));
typedef __bf16 v8bf  __attribute__((ext_vector_type(8)));
typedef float  v8f   __attribute__((ext_vector_type(8)));

static __device__ __forceinline__ v16bf cat8(v8bf lo, v8bf hi) {
  return __builtin_shufflevector(lo, hi, 0,1,2,3,4,5,6,7,8,9,10,11,12,13,14,15);
}

// async DMA one 16B chunk global -> LDS (gfx1250, ASYNCcnt-tracked)
static __device__ __forceinline__ void async_g2l_b128(unsigned lds_off,
                                                      const void* gaddr) {
  unsigned long long g = (unsigned long long)(uintptr_t)gaddr;
  asm volatile("global_load_async_to_lds_b128 %0, %1, off"
               :: "v"(lds_off), "v"(g) : "memory");
}
static __device__ __forceinline__ void wait_asynccnt0() {
  asm volatile("s_wait_asynccnt 0x0" ::: "memory");
}

// encode float so that unsigned compare == float compare (for atomicMax)
static __device__ __forceinline__ unsigned f32_ord_enc(float f) {
  unsigned u = __float_as_uint(f);
  return (u & 0x80000000u) ? ~u : (u | 0x80000000u);
}
static __device__ __forceinline__ float f32_ord_dec(unsigned u) {
  u = (u & 0x80000000u) ? (u & 0x7fffffffu) : ~u;
  return __uint_as_float(u);
}
#define ENC_NEG_INF 0x007FFFFFu   // f32_ord_enc(-inf)

// ---------------- utility kernels ----------------
__global__ void k_f32_to_bf16(const float* __restrict__ in, bf16_t* __restrict__ out, int n) {
  for (int i = blockIdx.x * blockDim.x + threadIdx.x; i < n; i += gridDim.x * blockDim.x)
    out[i] = (bf16_t)in[i];
}
__global__ void k_fill_f32(float* __restrict__ p, float v, int n) {
  for (int i = blockIdx.x * blockDim.x + threadIdx.x; i < n; i += gridDim.x * blockDim.x)
    p[i] = v;
}
__global__ void k_fill_u32(unsigned* __restrict__ p, unsigned v, int n) {
  for (int i = blockIdx.x * blockDim.x + threadIdx.x; i < n; i += gridDim.x * blockDim.x)
    p[i] = v;
}

// build transposed bf16 weight image (exact LDS layout incl. padding):
// wt[n*STR + k] = W[k*CC + n]  (n: 0..CC-1 output col, k: 0..K-1, pad k>=K -> 0)
__global__ void k_wt_bf16(const float* __restrict__ W, bf16_t* __restrict__ wt,
                          int K, int STR) {
  const int total = CC * STR;
  for (int i = blockIdx.x * blockDim.x + threadIdx.x; i < total; i += gridDim.x * blockDim.x) {
    int n = i / STR, k = i - n * STR;
    wt[i] = (k < K) ? (bf16_t)W[(size_t)k * CC + n] : (bf16_t)0.f;
  }
}

// ---------------- dual GEMM: xl = h@Wl, xr = h@Wr (bf16 WMMA, f32 acc) --------
// block = 128 threads (4 waves); wave owns 32 rows (two 16-row A tiles) so each
// LDS B fragment feeds 2 WMMAs. Weight LDS image DMA'd via async global->LDS.
__global__ void __launch_bounds__(128)
k_gemm_dual(const bf16_t* __restrict__ hb, const bf16_t* __restrict__ wt,
            float* __restrict__ xl, float* __restrict__ xr, int nrows) {
  extern __shared__ bf16_t smem[];                      // [2][CC][STRD]
  {
    const unsigned ldsbase = (unsigned)(uintptr_t)smem; // low 32b == LDS offset
    const char* gsrc = (const char*)wt;
    const int totalBytes = 2 * CC * STRD * (int)sizeof(bf16_t);   // 69632
    for (int off = threadIdx.x * 16; off < totalBytes; off += 128 * 16)
      async_g2l_b128(ldsbase + (unsigned)off, gsrc + off);
    wait_asynccnt0();
  }
  __syncthreads();
  const bf16_t* lwl = smem;
  const bf16_t* lwr = smem + CC * STRD;

  const int tid = threadIdx.x;
  const int wave = tid >> 5, lane = tid & 31;
  const int half = lane >> 4, l15 = lane & 15;
  const int rowbase = blockIdx.x * 128 + wave * 32;
  int r0 = rowbase + l15;      if (r0 >= nrows) r0 = nrows - 1;   // clamp only
  int r1 = rowbase + 16 + l15; if (r1 >= nrows) r1 = nrows - 1;
  const bf16_t* arow0 = hb + (long long)r0 * CC;
  const bf16_t* arow1 = hb + (long long)r1 * CC;

  v16bf a0[4], a1[4];                                   // A 16x32 frags, K=128
#pragma unroll
  for (int ks = 0; ks < 4; ++ks) {
    int k0 = ks * 32 + half * 8;
    a0[ks] = cat8(*(const v8bf*)(arow0 + k0), *(const v8bf*)(arow0 + k0 + 16));
    a1[ks] = cat8(*(const v8bf*)(arow1 + k0), *(const v8bf*)(arow1 + k0 + 16));
  }
  const bool full = (rowbase + 32) <= nrows;            // wave-uniform fast path

#pragma unroll
  for (int ct = 0; ct < 8; ++ct) {
    const int c = ct * 16 + l15;
    v8f al0 = {}, ar0 = {}, al1 = {}, ar1 = {};
#pragma unroll
    for (int ks = 0; ks < 4; ++ks) {
      int kk = ks * 32 + half * 8;
      v16bf bl = cat8(*(const v8bf*)(lwl + c * STRD + kk), *(const v8bf*)(lwl + c * STRD + kk + 16));
      v16bf br = cat8(*(const v8bf*)(lwr + c * STRD + kk), *(const v8bf*)(lwr + c * STRD + kk + 16));
      al0 = __builtin_amdgcn_wmma_f32_16x16x32_bf16(false, a0[ks], false, bl, (short)0, al0, false, false);
      ar0 = __builtin_amdgcn_wmma_f32_16x16x32_bf16(false, a0[ks], false, br, (short)0, ar0, false, false);
      al1 = __builtin_amdgcn_wmma_f32_16x16x32_bf16(false, a1[ks], false, bl, (short)0, al1, false, false);
      ar1 = __builtin_amdgcn_wmma_f32_16x16x32_bf16(false, a1[ks], false, br, (short)0, ar1, false, false);
    }
    if (full) {
#pragma unroll
      for (int v = 0; v < 8; ++v) {
        const long long row0 = rowbase + half * 8 + v;
        const long long row1 = row0 + 16;
        xl[row0 * CC + c] = al0[v];  xr[row0 * CC + c] = ar0[v];
        xl[row1 * CC + c] = al1[v];  xr[row1 * CC + c] = ar1[v];
      }
    } else {
#pragma unroll
      for (int v = 0; v < 8; ++v) {
        const int row0 = rowbase + half * 8 + v;
        const int row1 = row0 + 16;
        if (row0 < nrows) { xl[(long long)row0 * CC + c] = al0[v]; xr[(long long)row0 * CC + c] = ar0[v]; }
        if (row1 < nrows) { xl[(long long)row1 * CC + c] = al1[v]; xr[(long long)row1 * CC + c] = ar1[v]; }
      }
    }
  }
}

// ---------------- edge phase ----------------
// one wave32 per edge; lane covers 4 channels (float4)
__global__ void __launch_bounds__(256)
k_edge_e(const float* __restrict__ xl, const float* __restrict__ xr,
         const float* __restrict__ att, const int* __restrict__ src,
         const int* __restrict__ dst, float* __restrict__ ebuf,
         unsigned* __restrict__ emax) {
  const int eid = blockIdx.x * 8 + (threadIdx.x >> 5);
  const int lane = threadIdx.x & 31;
  if (eid >= ETOT) return;
  int s, d;
  if (eid < EE) { s = src[eid]; d = dst[eid]; } else { s = eid - EE; d = s; }
  const float4 a = *(const float4*)(xl + (long long)s * CC + lane * 4);
  const float4 b = *(const float4*)(xr + (long long)d * CC + lane * 4);
  const float4 w = *(const float4*)(att + lane * 4);
  float z0 = a.x + b.x; z0 = z0 >= 0.f ? z0 : NEG_SLOPE * z0;
  float z1 = a.y + b.y; z1 = z1 >= 0.f ? z1 : NEG_SLOPE * z1;
  float z2 = a.z + b.z; z2 = z2 >= 0.f ? z2 : NEG_SLOPE * z2;
  float z3 = a.w + b.w; z3 = z3 >= 0.f ? z3 : NEG_SLOPE * z3;
  float sum = z0 * w.x + z1 * w.y + z2 * w.z + z3 * w.w;
#pragma unroll
  for (int off = 16; off > 0; off >>= 1) sum += __shfl_xor(sum, off, 32);
  if (lane == 0) {
    ebuf[eid] = sum;
    atomicMax(emax + d, f32_ord_enc(sum));
  }
}

__global__ void __launch_bounds__(256)
k_edge_p(const float* __restrict__ ebuf, const unsigned* __restrict__ emax,
         const int* __restrict__ dst, float* __restrict__ pbuf,
         float* __restrict__ denom) {
  const int eid = blockIdx.x * blockDim.x + threadIdx.x;
  if (eid >= ETOT) return;
  const int d = (eid < EE) ? dst[eid] : (eid - EE);
  const float m = f32_ord_dec(emax[d]);
  const float p = __expf(ebuf[eid] - m);
  pbuf[eid] = p;
  atomicAdd(denom + d, p);
}

__global__ void __launch_bounds__(256)
k_edge_scatter(const float* __restrict__ pbuf, const float* __restrict__ denom,
               const float* __restrict__ xl, const int* __restrict__ src,
               const int* __restrict__ dst, float* __restrict__ hacc) {
  const int eid = blockIdx.x * 8 + (threadIdx.x >> 5);
  const int lane = threadIdx.x & 31;
  if (eid >= ETOT) return;
  int s, d;
  if (eid < EE) { s = src[eid]; d = dst[eid]; } else { s = eid - EE; d = s; }
  const float alpha = pbuf[eid] / denom[d];
  const float4 v = *(const float4*)(xl + (long long)s * CC + lane * 4);
  float* o = hacc + (long long)d * CC + lane * 4;
  atomicAdd(o + 0, alpha * v.x);
  atomicAdd(o + 1, alpha * v.y);
  atomicAdd(o + 2, alpha * v.z);
  atomicAdd(o + 3, alpha * v.w);
}

__global__ void k_finalize_h(float* __restrict__ h, const float* __restrict__ bias,
                             int relu, int n) {
  for (int i = blockIdx.x * blockDim.x + threadIdx.x; i < n; i += gridDim.x * blockDim.x) {
    float v = h[i] + bias[i & (CC - 1)];
    if (relu) v = v > 0.f ? v : 0.f;
    h[i] = v;
  }
}

// ---------------- scorer ----------------
__global__ void __launch_bounds__(256)
k_gather_pairs(const float* __restrict__ h, const int* __restrict__ tp,
               const int* __restrict__ fp, float* __restrict__ outpn,
               bf16_t* __restrict__ vb) {
  const int r = blockIdx.x;          // 0 .. 2P-1
  const int c = threadIdx.x;         // 0 .. 255
  const int* pr = (r < PP) ? (tp + r * 2) : (fp + (r - PP) * 2);
  const int node = (c < CC) ? pr[0] : pr[1];
  const float v = h[(long long)node * CC + (c & (CC - 1))];
  outpn[(long long)r * 256 + c] = v;
  vb[(long long)r * 256 + c] = (bf16_t)v;
}

// hid = relu(v @ W1 + b1); v: 2P x 256 (bf16), W1 image pre-transposed, K=256
__global__ void __launch_bounds__(128)
k_gemm_scorer(const bf16_t* __restrict__ vb, const bf16_t* __restrict__ wt1,
              const float* __restrict__ b1, float* __restrict__ hid) {
  extern __shared__ bf16_t smem[];                      // [CC][STRS]
  {
    const unsigned ldsbase = (unsigned)(uintptr_t)smem;
    const char* gsrc = (const char*)wt1;
    const int totalBytes = CC * STRS * (int)sizeof(bf16_t);       // 67584
    for (int off = threadIdx.x * 16; off < totalBytes; off += 128 * 16)
      async_g2l_b128(ldsbase + (unsigned)off, gsrc + off);
    wait_asynccnt0();
  }
  __syncthreads();
  const bf16_t* lw = smem;

  const int tid = threadIdx.x;
  const int wave = tid >> 5, lane = tid & 31;
  const int half = lane >> 4, l15 = lane & 15;
  const int rowbase = blockIdx.x * 64 + wave * 16;      // 2P % 64 == 0, no guard
  const bf16_t* arow = vb + (long long)(rowbase + l15) * 256;

  v16bf afr[8];
#pragma unroll
  for (int ks = 0; ks < 8; ++ks) {
    int k0 = ks * 32 + half * 8;
    afr[ks] = cat8(*(const v8bf*)(arow + k0), *(const v8bf*)(arow + k0 + 16));
  }
#pragma unroll
  for (int ct = 0; ct < 8; ++ct) {
    const int c = ct * 16 + l15;
    v8f acc = {};
#pragma unroll
    for (int ks = 0; ks < 8; ++ks) {
      int kk = ks * 32 + half * 8;
      v16bf b = cat8(*(const v8bf*)(lw + c * STRS + kk), *(const v8bf*)(lw + c * STRS + kk + 16));
      acc = __builtin_amdgcn_wmma_f32_16x16x32_bf16(false, afr[ks], false, b, (short)0, acc, false, false);
    }
    const float bc = b1[c];
#pragma unroll
    for (int v = 0; v < 8; ++v) {
      const int row = rowbase + half * 8 + v;
      float val = acc[v] + bc;
      hid[(long long)row * CC + c] = val > 0.f ? val : 0.f;
    }
  }
}

__global__ void __launch_bounds__(256)
k_probs(const float* __restrict__ hid, const float* __restrict__ W2,
        const float* __restrict__ b2, float* __restrict__ probs,
        float* __restrict__ lossacc) {
  const int r = blockIdx.x * 8 + (threadIdx.x >> 5);
  const int lane = threadIdx.x & 31;
  if (r >= 2 * PP) return;
  const float4 hv = *(const float4*)(hid + (long long)r * CC + lane * 4);
  const float4 wv = *(const float4*)(W2 + lane * 4);
  float s = hv.x * wv.x + hv.y * wv.y + hv.z * wv.z + hv.w * wv.w;
#pragma unroll
  for (int off = 16; off > 0; off >>= 1) s += __shfl_xor(s, off, 32);
  if (lane == 0) {
    const float logit = s + b2[0];
    const float pr = 1.f / (1.f + __expf(-logit));
    probs[r] = pr;
    const float pcl = fminf(fmaxf(pr, 1e-7f), 1.f - 1e-7f);
    const float term = (r < PP) ? __logf(pcl) : log1pf(-pcl);
    atomicAdd(lossacc, term);
  }
}

__global__ void k_loss_fin(const float* __restrict__ lossacc, float* __restrict__ out) {
  out[0] = -lossacc[0] * (1.0f / (2.0f * (float)PP));
}

// ---------------- host launcher ----------------
extern "C" void kernel_launch(void* const* d_in, const int* in_sizes, int n_in,
                              void* d_out, int out_size, void* d_ws, size_t ws_size,
                              hipStream_t stream) {
  const float* x    = (const float*)d_in[0];
  const int*   ei   = (const int*)  d_in[1];
  const int*   tp   = (const int*)  d_in[2];
  const int*   fp   = (const int*)  d_in[3];
  const float* Wl   = (const float*)d_in[4];   // [L,C,C]
  const float* Wr   = (const float*)d_in[5];
  const float* attw = (const float*)d_in[6];   // [L,C]
  const float* bias = (const float*)d_in[7];   // [L,C]
  const float* W1   = (const float*)d_in[8];   // [2C,C]
  const float* b1   = (const float*)d_in[9];
  const float* W2   = (const float*)d_in[10];  // [C,1]
  const float* b2   = (const float*)d_in[11];
  float* out = (float*)d_out;

  const int* srcp = ei;
  const int* dstp = ei + EE;

  // ---- workspace carve-up (256B aligned segments) ----
  char* ws = (char*)d_ws;
  size_t cur = 0;
  auto take = [&](size_t bytes) { void* p = ws + cur; cur += (bytes + 255) & ~(size_t)255; return p; };
  float*    h1    = (float*)   take((size_t)NN * CC * 4);
  float*    h2    = (float*)   take((size_t)NN * CC * 4);
  float*    xl    = (float*)   take((size_t)NN * CC * 4);
  float*    xr    = (float*)   take((size_t)NN * CC * 4);
  float*    hid   = (float*)   take((size_t)2 * PP * CC * 4);
  float*    ebuf  = (float*)   take((size_t)ETOT * 4);
  float*    pbuf  = (float*)   take((size_t)ETOT * 4);
  float*    denom = (float*)   take((size_t)NN * 4);
  unsigned* emax  = (unsigned*)take((size_t)NN * 4);
  float*    lacc  = (float*)   take(256);
  bf16_t*   hb    = (bf16_t*)  take((size_t)NN * CC * 2);
  bf16_t*   vb    = (bf16_t*)  take((size_t)2 * PP * 256 * 2);
  bf16_t*   wtd0  = (bf16_t*)  take((size_t)2 * CC * STRD * 2);  // layer0 Wl^T|Wr^T image
  bf16_t*   wtd1  = (bf16_t*)  take((size_t)2 * CC * STRD * 2);  // layer1
  bf16_t*   wt1i  = (bf16_t*)  take((size_t)CC * STRS * 2);      // W1^T image
  (void)ws_size; (void)n_in; (void)in_sizes; (void)out_size;

  const int NC = NN * CC;
  const int gemmBlocks   = (NN + 127) / 128;
  const int gemmShmem    = 2 * CC * STRD * (int)sizeof(bf16_t);   // 69632 B
  const int scorerShmem  = CC * STRS * (int)sizeof(bf16_t);       // 67584 B
  const int edgeWaveBlks = (ETOT + 7) / 8;
  const int edgeThrBlks  = (ETOT + 255) / 256;

  // build transposed bf16 weight images once (tiny)
  k_wt_bf16<<<64, 256, 0, stream>>>(Wl,              wtd0,             CC,  STRD);
  k_wt_bf16<<<64, 256, 0, stream>>>(Wr,              wtd0 + CC * STRD, CC,  STRD);
  k_wt_bf16<<<64, 256, 0, stream>>>(Wl + CC * CC,    wtd1,             CC,  STRD);
  k_wt_bf16<<<64, 256, 0, stream>>>(Wr + CC * CC,    wtd1 + CC * STRD, CC,  STRD);
  k_wt_bf16<<<64, 256, 0, stream>>>(W1,              wt1i,             2*CC, STRS);

  for (int l = 0; l < 2; ++l) {
    const float* hin  = (l == 0) ? x  : h1;
    float*       hacc = (l == 0) ? h1 : h2;   // accumulate, then finalize in place
    const bf16_t* wtd = (l == 0) ? wtd0 : wtd1;

    // dense phase: bf16 convert + dual WMMA GEMM (weights DMA'd async to LDS)
    k_f32_to_bf16<<<4096, 256, 0, stream>>>(hin, hb, NC);
    k_gemm_dual<<<gemmBlocks, 128, gemmShmem, stream>>>(hb, wtd, xl, xr, NN);

    // init per-node state
    k_fill_f32<<<1024, 256, 0, stream>>>(hacc, 0.f, NC);
    k_fill_f32<<<256, 256, 0, stream>>>(denom, 0.f, NN);
    k_fill_u32<<<256, 256, 0, stream>>>(emax, ENC_NEG_INF, NN);

    // sparse phase: attention logits -> softmax -> aggregate
    k_edge_e<<<edgeWaveBlks, 256, 0, stream>>>(xl, xr, attw + (size_t)l * CC,
                                               srcp, dstp, ebuf, emax);
    k_edge_p<<<edgeThrBlks, 256, 0, stream>>>(ebuf, emax, dstp, pbuf, denom);
    k_edge_scatter<<<edgeWaveBlks, 256, 0, stream>>>(pbuf, denom, xl, srcp, dstp, hacc);
    k_finalize_h<<<4096, 256, 0, stream>>>(hacc, bias + (size_t)l * CC,
                                           (l == 0) ? 1 : 0, NC);
  }

  // scorer: gather pairs (writes pos||neg into d_out), WMMA MLP, sigmoid + BCE
  k_gather_pairs<<<2 * PP, 256, 0, stream>>>(h2, tp, fp, out, vb);
  k_gemm_scorer<<<(2 * PP) / 64, 128, scorerShmem, stream>>>(vb, wt1i, b1, hid);
  k_fill_f32<<<1, 32, 0, stream>>>(lacc, 0.f, 1);
  k_probs<<<(2 * PP + 7) / 8, 256, 0, stream>>>(hid, W2, b2,
                                                out + (size_t)2 * PP * 256, lacc);
  k_loss_fin<<<1, 1, 0, stream>>>(lacc, out + (size_t)2 * PP * 256 + 2 * PP);
}
